// Attention_2216203125487
// MI455X (gfx1250) — matrix operand order repaired and checked by reading, not compiled, not run on hardware
//
#include <hip/hip_runtime.h>
#include <hip/hip_bf16.h>
#include <math.h>

// ---------------------------------------------------------------------------
// Types for CDNA5 WMMA (wave32): v_wmma_f32_16x16x32_bf16
// ---------------------------------------------------------------------------
typedef __bf16 bf16_t;
typedef __attribute__((ext_vector_type(16))) __bf16 v16bf;
typedef __attribute__((ext_vector_type(8)))  float  v8f;
typedef __attribute__((ext_vector_type(4)))  unsigned v4u;
typedef __attribute__((ext_vector_type(8)))  int      v8i;
typedef __attribute__((ext_vector_type(4)))  int      v4i;

// f32 -> bf16: native v_cvt(_pk)_bf16_f32 on gfx1250
__device__ __forceinline__ bf16_t f2bf(float f)   { return (bf16_t)f; }
__device__ __forceinline__ bf16_t to_bf(float f)  { return (bf16_t)f; }
__device__ __forceinline__ bf16_t to_bf(bf16_t h) { return h; }

#if __has_builtin(__builtin_amdgcn_tensor_load_to_lds)
#define HAVE_TDM 1
#else
#define HAVE_TDM 0
#endif

// ---------------------------------------------------------------------------
// Fallback async global->LDS 16B copy (gfx1250: global_load_async_to_lds_b128)
// LDS address = low 32 bits of the generic pointer (ISA 10.2).
// ---------------------------------------------------------------------------
__device__ __forceinline__ void g2l_16(const bf16_t* g, bf16_t* l) {
#if __has_builtin(__builtin_amdgcn_load_to_lds)
  __builtin_amdgcn_load_to_lds(
      (void*)g,
      (__attribute__((address_space(3))) void*)(unsigned)(uintptr_t)l,
      16, 0, 0);
#else
  struct B16 { unsigned x[4]; };
  *(B16*)l = *(const B16*)g;
#endif
}

__device__ __forceinline__ void wait_async_copies() {
#if __has_builtin(__builtin_amdgcn_load_to_lds)
#if __has_builtin(__builtin_amdgcn_s_wait_asynccnt)
  __builtin_amdgcn_s_wait_asynccnt(0);
#else
  asm volatile("s_wait_asynccnt 0" ::: "memory");
#endif
#endif
}

__device__ __forceinline__ void wait_tdm() {
#if __has_builtin(__builtin_amdgcn_s_wait_tensorcnt)
  __builtin_amdgcn_s_wait_tensorcnt(0);
#else
  asm volatile("s_wait_tensorcnt 0" ::: "memory");
#endif
}

// ---------------------------------------------------------------------------
// TDM: issue one tensor_load_to_lds for a (up to) 2D tile.
// D# bit layout per ISA 8.3/8.4:
//  g0: [1:0]=count(1) | [63:32]=lds_addr | [120:64]=global_addr | [127:126]=2
//  g1: [17:16]=data_size | [79:48]=tensor_dim0 | [111:80]=tensor_dim1
//      [127:112]=tile_dim0 | [143:128]=tile_dim1 | [207:160]=tensor_dim0_stride
// data_size_log: 0=1B,1=2B,2=4B,3=8B; dims/strides in data_size units.
// ---------------------------------------------------------------------------
__device__ __forceinline__ void tdm_load(unsigned lds_off, const void* gaddr,
                                         unsigned data_size_log,
                                         unsigned tile_d0, unsigned tile_d1,
                                         unsigned tensor_d0, unsigned tensor_d1,
                                         unsigned long long stride0) {
#if HAVE_TDM
  unsigned long long ga = (unsigned long long)(uintptr_t)gaddr;
  v4u g0;
  g0[0] = 1u;                                        // count=1, user desc
  g0[1] = lds_off;                                   // lds_addr
  g0[2] = (unsigned)ga;                              // global_addr[31:0]
  g0[3] = (unsigned)((ga >> 32) & 0x01FFFFFFu) | (2u << 30);  // addr hi | type=2
  v8i g1;
  g1[0] = (int)(data_size_log << 16);                // wg_mask=0, data_size
  g1[1] = (int)((tensor_d0 & 0xFFFFu) << 16);        // tensor_dim0 lo
  g1[2] = (int)(((tensor_d0 >> 16) & 0xFFFFu) | ((tensor_d1 & 0xFFFFu) << 16));
  g1[3] = (int)(((tensor_d1 >> 16) & 0xFFFFu) | ((tile_d0 & 0xFFFFu) << 16));
  g1[4] = (int)(tile_d1 & 0xFFFFu);                  // tile_dim1 (tile_dim2=0)
  g1[5] = (int)(stride0 & 0xFFFFFFFFu);              // dim0 stride lo
  g1[6] = (int)((stride0 >> 32) & 0xFFFFu);          // dim0 stride hi
  g1[7] = 0;
  v4i z4 = {0, 0, 0, 0};
#if __clang_major__ >= 23
  v8i z8 = {0, 0, 0, 0, 0, 0, 0, 0};
  __builtin_amdgcn_tensor_load_to_lds(g0, g1, z4, z4, z8, 0);
#else
  __builtin_amdgcn_tensor_load_to_lds(g0, g1, z4, z4, 0);
#endif
#endif
}

// ---------------------------------------------------------------------------
// WMMA operand loaders (layouts per ISA 7.12.2).
// ---------------------------------------------------------------------------
template <typename AT>
__device__ __forceinline__ v16bf load_a(const AT* A, int lda, int row0, int k0, int lane) {
  int r  = row0 + (lane & 15);
  int kh = (lane >> 4) & 1;
  const AT* p0 = A + (size_t)r * lda + k0 + kh * 8;
  v16bf v;
#pragma unroll
  for (int i = 0; i < 8; ++i) v[i]     = to_bf(p0[i]);
#pragma unroll
  for (int i = 0; i < 8; ++i) v[8 + i] = to_bf(p0[16 + i]);
  return v;
}

__device__ __forceinline__ v16bf load_bt(const bf16_t* Bt, int ldk, int k0, int col0, int lane) {
  const bf16_t* p = Bt + (size_t)(col0 + (lane & 15)) * ldk + k0 + ((lane >> 4) & 1) * 8;
  v16bf v;
#pragma unroll
  for (int i = 0; i < 8; ++i) v[i]     = p[i];
#pragma unroll
  for (int i = 0; i < 8; ++i) v[8 + i] = p[16 + i];
  return v;
}

#define WMMA_BF16(a, b, c) \
  __builtin_amdgcn_wmma_f32_16x16x32_bf16(false, (a), false, (b), (short)0, (c), false, false)

// ---------------------------------------------------------------------------
// f32 -> bf16 transpose-convert for weights: wt[n*K + k] = w[k*N + n]
// ---------------------------------------------------------------------------
__global__ void convert_transpose(const float* __restrict__ w, bf16_t* __restrict__ wt,
                                  int K, int N) {
  int idx = blockIdx.x * blockDim.x + threadIdx.x;
  if (idx >= K * N) return;
  int k = idx / N, n = idx - k * N;
  wt[(size_t)n * K + k] = f2bf(w[idx]);
}

// ---------------------------------------------------------------------------
// Generic GEMM (K = 256): C[M x N] = A[M x K] x Bt[N x K] (+bias).
// 256 threads = 8 waves; wave owns 16 rows x 64 cols (4 D tiles) -> WG does
// 128 rows x 64 cols.  The contiguous 32KB B panel is staged with ONE TDM
// descriptor (tensor_load_to_lds, TENSORcnt), amortized over 8 waves.
// ---------------------------------------------------------------------------
template <typename AT>
__global__ __launch_bounds__(256) void gemm_abt(const AT* __restrict__ A,
                                                const bf16_t* __restrict__ Bt,
                                                float* Cf, bf16_t* Cb,
                                                const float* __restrict__ bias,
                                                int M, int Ncols,
                                                int lda, int ldc) {
  constexpr int K = 256;
  __shared__ __align__(16) bf16_t bpanel[64 * K];   // 32 KB

  int lane = threadIdx.x & 31;
  int wave = threadIdx.x >> 5;
  int row0 = blockIdx.y * 128 + wave * 16;
  int col0 = blockIdx.x * 64;

  const bf16_t* bsrc = Bt + (size_t)col0 * K;
#if HAVE_TDM
  if (wave == 0) {
    // 1D copy: 8192 dwords, tile==tensor
    tdm_load((unsigned)(uintptr_t)&bpanel[0], bsrc, /*4B*/2,
             8192, 1, 8192, 1, 8192);
    wait_tdm();
  }
#else
#pragma unroll
  for (int c = 0; c < 8; ++c) {
    int e = (c * 256 + (int)threadIdx.x) * 8;       // 16B chunks
    g2l_16(bsrc + e, &bpanel[e]);
  }
  wait_async_copies();
#endif
  __syncthreads();

  v8f acc[4];
#pragma unroll
  for (int t = 0; t < 4; ++t)
#pragma unroll
    for (int i = 0; i < 8; ++i) acc[t][i] = 0.0f;

#pragma unroll 2
  for (int k0 = 0; k0 < K; k0 += 32) {
    v16bf a = load_a(A, lda, row0, k0, lane);
#pragma unroll
    for (int t = 0; t < 4; ++t) {
      v16bf b = load_bt(bpanel, K, k0, t * 16, lane);   // ds_load_b128 x2
      acc[t] = WMMA_BF16(a, b, acc[t]);
    }
  }

  int rr = (lane >> 4) * 8;
  int cc = lane & 15;
#pragma unroll
  for (int t = 0; t < 4; ++t) {
    int c = col0 + t * 16 + cc;
    float bv = bias ? bias[c] : 0.0f;
#pragma unroll
    for (int i = 0; i < 8; ++i) {
      size_t off = (size_t)(row0 + rr + i) * ldc + c;
      float vv = acc[t][i] + bv;
      if (Cf) Cf[off] = vv; else Cb[off] = f2bf(vv);
    }
  }
}

// ---------------------------------------------------------------------------
// SR conv (kernel=stride=4, VALID) as gathered GEMM:
// rows = B*M (2048 tokens), K = 4096 = (py,px,c), cols = 256.
// K chunk of 32 always lies inside one (py,px) channel block -> A load is
// two contiguous 32B f32 reads + packed cvt.  B panel (512KB) stays in L2.
// ---------------------------------------------------------------------------
__global__ __launch_bounds__(128) void sr_conv(const float* __restrict__ x,
                                               const bf16_t* __restrict__ srkt,
                                               float* __restrict__ xr) {
  int lane = threadIdx.x & 31;
  int wave = threadIdx.x >> 5;
  int row0 = blockIdx.y * 64 + wave * 16;
  int col0 = blockIdx.x * 64;
  int b  = row0 >> 8;
  int my = (row0 >> 4) & 15;
  int mx = lane & 15;
  int kh = (lane >> 4) & 1;

  v8f acc[4];
#pragma unroll
  for (int t = 0; t < 4; ++t)
#pragma unroll
    for (int i = 0; i < 8; ++i) acc[t][i] = 0.0f;

  for (int k0 = 0; k0 < 4096; k0 += 32) {
    int blk = k0 >> 8;                 // py*4+px
    int py = blk >> 2, px = blk & 3;
    int c0 = k0 & 255;
    const float* p0 = x + ((((size_t)b * 64 + my * 4 + py) * 64) + mx * 4 + px) * 256
                        + c0 + kh * 8;
    v16bf a;
#pragma unroll
    for (int i = 0; i < 8; ++i) a[i]     = f2bf(p0[i]);
#pragma unroll
    for (int i = 0; i < 8; ++i) a[8 + i] = f2bf(p0[16 + i]);
#pragma unroll
    for (int t = 0; t < 4; ++t) {
      v16bf bm = load_bt(srkt, 4096, k0, col0 + t * 16, lane);
      acc[t] = WMMA_BF16(a, bm, acc[t]);
    }
  }

  int rr = (lane >> 4) * 8;
  int cc = lane & 15;
#pragma unroll
  for (int t = 0; t < 4; ++t)
#pragma unroll
    for (int i = 0; i < 8; ++i)
      xr[(size_t)(row0 + rr + i) * 256 + col0 + t * 16 + cc] = acc[t][i];
}

// ---------------------------------------------------------------------------
// LayerNorm over dim=256 (adds sr_bias first), bf16 out. One block per row.
// ---------------------------------------------------------------------------
__global__ __launch_bounds__(256) void ln_kernel(const float* __restrict__ xr,
                                                 const float* __restrict__ sbias,
                                                 const float* __restrict__ gamma,
                                                 const float* __restrict__ beta,
                                                 bf16_t* __restrict__ out) {
  int row = blockIdx.x;
  int c   = threadIdx.x;
  float v = xr[(size_t)row * 256 + c] + sbias[c];
  __shared__ float red[256];
  red[c] = v; __syncthreads();
  for (int s = 128; s > 0; s >>= 1) { if (c < s) red[c] += red[c + s]; __syncthreads(); }
  float mu = red[0] * (1.0f / 256.0f); __syncthreads();
  float d = v - mu;
  red[c] = d * d; __syncthreads();
  for (int s = 128; s > 0; s >>= 1) { if (c < s) red[c] += red[c + s]; __syncthreads(); }
  float var = red[0] * (1.0f / 256.0f);
  float y = d * rsqrtf(var + 1e-3f) * gamma[c] + beta[c];
  out[(size_t)row * 256 + c] = f2bf(y);
}

// ---------------------------------------------------------------------------
// Transpose V half of KV into [b*8+head][d(32)][key(256)].
// ---------------------------------------------------------------------------
__global__ void v_transpose(const bf16_t* __restrict__ kvb, bf16_t* __restrict__ vtb) {
  int idx = blockIdx.x * blockDim.x + threadIdx.x;   // B*M*256
  if (idx >= 8 * 256 * 256) return;
  int j = idx & 255;
  int m = (idx >> 8) & 255;
  int b = idx >> 16;
  int head = j >> 5, d = j & 31;
  vtb[(((size_t)(b * 8 + head) * 32) + d) * 256 + m] =
      kvb[((size_t)(b * 256) + m) * 512 + 256 + j];
}

// ---------------------------------------------------------------------------
// Fused attention, 8 waves = 128 query rows of one (b, head) per WG.
// K panel staged via a 2D TDM descriptor (256 keys x 32 dims, row stride 512),
// V panel (transposed, contiguous 16KB) via a 1D TDM descriptor.
// QK^T: 16 wmmas/wave (K=hd=32); in-register full-row softmax over M=256;
// P -> LDS (D->A layout transpose); PV: 16 wmmas/wave from LDS.
// Never materializes the 134MB attention matrix in HBM.
// ---------------------------------------------------------------------------
__global__ __launch_bounds__(256) void attn_kernel(const bf16_t* __restrict__ qb,
                                                   const bf16_t* __restrict__ kvb,
                                                   const bf16_t* __restrict__ vtb,
                                                   bf16_t* __restrict__ ob) {
  constexpr int N = 4096, Mk = 256, DIM = 256, HD = 32;
  int lane = threadIdx.x & 31;
  int wave = threadIdx.x >> 5;
  int wg    = blockIdx.x;          // B*h*(N/128) = 2048
  int ntile = wg & 31;
  int head  = (wg >> 5) & 7;
  int b     = wg >> 8;
  int n0    = ntile * 128 + wave * 16;

  __shared__ __align__(16) bf16_t plds[8][16][Mk + 8];  // 66 KB
  __shared__ __align__(16) bf16_t klds[Mk * HD];        // 16 KB [key][d]
  __shared__ __align__(16) bf16_t vlds[HD * Mk];        // 16 KB [d][key]

  const bf16_t* ksrc = kvb + (size_t)b * Mk * 512 + head * HD;
  const bf16_t* vsrc = vtb + (size_t)(b * 8 + head) * HD * Mk;
#if HAVE_TDM
  if (wave == 0) {
    // K: 2D tile, 2B elems: 32 per row, 256 rows, tensor row stride 512
    tdm_load((unsigned)(uintptr_t)&klds[0], ksrc, /*2B*/1, HD, Mk, HD, Mk, 512);
    // V: contiguous 16KB = 4096 dwords
    tdm_load((unsigned)(uintptr_t)&vlds[0], vsrc, /*4B*/2, 4096, 1, 4096, 1, 4096);
    wait_tdm();
  }
#else
  {
    int t = threadIdx.x;                       // 256 threads
#pragma unroll
    for (int c = 0; c < 4; ++c)                // K: one 64B row per thread
      g2l_16(ksrc + (size_t)t * 512 + c * 8, &klds[t * HD + c * 8]);
#pragma unroll
    for (int c = 0; c < 4; ++c) {              // V: 1024 16B chunks
      int e = (c * 256 + t) * 8;
      g2l_16(vsrc + e, &vlds[e]);
    }
    wait_async_copies();
  }
#endif
  __syncthreads();

  int qr = lane & 15;
  int kh = (lane >> 4) & 1;

  // Q A-tile (K = full head dim 32)
  const bf16_t* qp = qb + ((size_t)b * N + n0 + qr) * DIM + head * HD;
  v16bf qa;
#pragma unroll
  for (int i = 0; i < 8; ++i) qa[i]     = qp[kh * 8 + i];
#pragma unroll
  for (int i = 0; i < 8; ++i) qa[8 + i] = qp[16 + kh * 8 + i];

  // scores: 16 tiles of 16 keys = all 256 keys; K B-tiles from LDS
  v8f acc[16];
#pragma unroll
  for (int t = 0; t < 16; ++t) {
    const bf16_t* kp = &klds[(t * 16 + (lane & 15)) * HD + kh * 8];
    v16bf kb;
#pragma unroll
    for (int i = 0; i < 8; ++i) kb[i]     = kp[i];
#pragma unroll
    for (int i = 0; i < 8; ++i) kb[8 + i] = kp[16 + i];
    v8f z;
#pragma unroll
    for (int i = 0; i < 8; ++i) z[i] = 0.0f;
    acc[t] = WMMA_BF16(qa, kb, z);
  }

  const float scale = 0.17677669529663687f;  // 32^-0.5
  float rinv[8];
#pragma unroll
  for (int i = 0; i < 8; ++i) {
    float m = -3.0e38f;
#pragma unroll
    for (int t = 0; t < 16; ++t) m = fmaxf(m, acc[t][i]);
    for (int x = 1; x < 16; x <<= 1) m = fmaxf(m, __shfl_xor(m, x, 32));
    float sum = 0.0f;
#pragma unroll
    for (int t = 0; t < 16; ++t) {
      float e = __expf((acc[t][i] - m) * scale);
      acc[t][i] = e;
      sum += e;
    }
    for (int x = 1; x < 16; x <<= 1) sum += __shfl_xor(sum, x, 32);
    rinv[i] = 1.0f / sum;
  }

  // stage normalized P into LDS (row-major bf16)
  int prow0 = (lane >> 4) * 8;
  int pc    = lane & 15;
#pragma unroll
  for (int t = 0; t < 16; ++t)
#pragma unroll
    for (int i = 0; i < 8; ++i)
      plds[wave][prow0 + i][t * 16 + pc] = f2bf(acc[t][i] * rinv[i]);
  __syncthreads();

  // PV: accumulate over 8 K-chunks of 32 keys; 2 output col tiles (hd=32)
  v8f oacc[2];
#pragma unroll
  for (int oc = 0; oc < 2; ++oc)
#pragma unroll
    for (int i = 0; i < 8; ++i) oacc[oc][i] = 0.0f;

#pragma unroll 2
  for (int kc = 0; kc < 8; ++kc) {
    const bf16_t* pp = &plds[wave][qr][kc * 32];
    v16bf pa;
#pragma unroll
    for (int i = 0; i < 8; ++i) pa[i]     = pp[kh * 8 + i];
#pragma unroll
    for (int i = 0; i < 8; ++i) pa[8 + i] = pp[16 + kh * 8 + i];
#pragma unroll
    for (int oc = 0; oc < 2; ++oc) {
      const bf16_t* vp = &vlds[(oc * 16 + (lane & 15)) * Mk + kc * 32 + kh * 8];
      v16bf vb;
#pragma unroll
      for (int i = 0; i < 8; ++i) vb[i]     = vp[i];
#pragma unroll
      for (int i = 0; i < 8; ++i) vb[8 + i] = vp[16 + i];
      oacc[oc] = WMMA_BF16(pa, vb, oacc[oc]);
    }
  }

#pragma unroll
  for (int oc = 0; oc < 2; ++oc)
#pragma unroll
    for (int i = 0; i < 8; ++i)
      ob[((size_t)b * N + n0 + prow0 + i) * DIM + head * HD + oc * 16 + pc] =
          f2bf(oacc[oc][i]);
}

// ---------------------------------------------------------------------------
// Host side
// ---------------------------------------------------------------------------
extern "C" void kernel_launch(void* const* d_in, const int* in_sizes, int n_in,
                              void* d_out, int out_size, void* d_ws, size_t ws_size,
                              hipStream_t stream) {
  const float* x   = (const float*)d_in[0];   // [8,4096,256]
  const float* Wq  = (const float*)d_in[1];   // [256,256]
  const float* Wkv = (const float*)d_in[2];   // [256,512]
  const float* srk = (const float*)d_in[3];   // [4,4,256,256] -> [4096,256]
  const float* sb  = (const float*)d_in[4];   // [256]
  const float* gam = (const float*)d_in[5];   // [256]
  const float* bet = (const float*)d_in[6];   // [256]
  const float* Wp  = (const float*)d_in[7];   // [256,256]
  const float* bp  = (const float*)d_in[8];   // [256]
  float* out = (float*)d_out;                 // [8,4096,256]

  char* ws = (char*)d_ws;
  size_t off = 0;
  auto alloc = [&](size_t bytes) -> void* {
    void* p = ws + off;
    off = (off + bytes + 255) & ~(size_t)255;
    return p;
  };
  bf16_t* qb   = (bf16_t*)alloc((size_t)32768 * 256 * 2);
  bf16_t* ob   = (bf16_t*)alloc((size_t)32768 * 256 * 2);
  bf16_t* Wqt  = (bf16_t*)alloc((size_t)256 * 256 * 2);
  bf16_t* Wkvt = (bf16_t*)alloc((size_t)512 * 256 * 2);
  bf16_t* srkt = (bf16_t*)alloc((size_t)256 * 4096 * 2);
  bf16_t* Wpt  = (bf16_t*)alloc((size_t)256 * 256 * 2);
  float*  xr   = (float*) alloc((size_t)2048 * 256 * 4);
  bf16_t* xnb  = (bf16_t*)alloc((size_t)2048 * 256 * 2);
  bf16_t* kvb  = (bf16_t*)alloc((size_t)2048 * 512 * 2);
  bf16_t* vtb  = (bf16_t*)alloc((size_t)2048 * 256 * 2);
  (void)ws_size; (void)in_sizes; (void)n_in; (void)out_size;

  // 1) weight transpose-converts (tiny)
  convert_transpose<<<(256 * 256 + 255) / 256, 256, 0, stream>>>(Wq,  Wqt,  256,  256);
  convert_transpose<<<(256 * 512 + 255) / 256, 256, 0, stream>>>(Wkv, Wkvt, 256,  512);
  convert_transpose<<<(4096 * 256 + 255) / 256, 256, 0, stream>>>(srk, srkt, 4096, 256);
  convert_transpose<<<(256 * 256 + 255) / 256, 256, 0, stream>>>(Wp,  Wpt,  256,  256);

  // 2) Q = x @ Wq  (f32 A converted inline, bf16 out)  M=32768 K=256 N=256
  gemm_abt<float><<<dim3(4, 256), 256, 0, stream>>>(x, Wqt, nullptr, qb, nullptr,
                                                    32768, 256, 256, 256);

  // 3) SR conv as GEMM: xr[2048,256]
  sr_conv<<<dim3(4, 32), 128, 0, stream>>>(x, srkt, xr);

  // 4) LayerNorm (+sr_bias) -> bf16
  ln_kernel<<<2048, 256, 0, stream>>>(xr, sb, gam, bet, xnb);

  // 5) KV = xn @ Wkv  M=2048 K=256 N=512
  gemm_abt<bf16_t><<<dim3(8, 16), 256, 0, stream>>>(xnb, Wkvt, nullptr, kvb, nullptr,
                                                    2048, 512, 256, 512);

  // 6) transpose V for contiguous PV B-tiles
  v_transpose<<<(8 * 256 * 256 + 255) / 256, 256, 0, stream>>>(kvb, vtb);

  // 7) fused attention (QK^T + softmax + PV), bf16 out
  attn_kernel<<<2048, 256, 0, stream>>>(qb, kvb, vtb, ob);

  // 8) out = attn_out @ Wp + bp  (f32 final output)
  gemm_abt<bf16_t><<<dim3(4, 256), 256, 0, stream>>>(ob, Wpt, out, nullptr, bp,
                                                     32768, 256, 256, 256);
}
